// IGLOO1Dseq_KERNEL_WK2style_39006892982784
// MI455X (gfx1250) — matrix operand order repaired and checked by reading, not compiled, not run on hardware
//
#include <hip/hip_runtime.h>

typedef __attribute__((ext_vector_type(16))) _Float16 v16h;
typedef __attribute__((ext_vector_type(8)))  _Float16 v8h;
typedef __attribute__((ext_vector_type(8)))  float    v8f;

#define BB  2
#define TT  4096
#define CC  512
#define PP  64
#define PSS 16
#define CSS 5
#define MBB 128
#define FDD 128

// ===================== WMMA fragment helpers (ISA 7.12.2 layouts) =====================

// Fragment from packed f16 [tile][lane][16] (works for global packed weights and for
// LDS-staged packed A tiles): two contiguous 16-byte loads per lane.
__device__ __forceinline__ v16h ld_frag_pk(const _Float16* pk, int tile, int lane) {
  const v8h* p = (const v8h*)(pk + ((size_t)tile * 32 + lane) * 16);
  return __builtin_shufflevector(p[0], p[1],
                                 0,1,2,3,4,5,6,7,8,9,10,11,12,13,14,15);
}

// A fragment from 16-byte aligned LDS f16 row-major tile, row stride `stride` halves.
__device__ __forceinline__ v16h ldA_lds(const _Float16* s, int stride, int kk, int lane) {
  int m  = lane & 15;
  int kb = (lane >> 4) * 8;
  const v8h* p0 = (const v8h*)(s + m * stride + kk + kb);
  const v8h* p1 = (const v8h*)(s + m * stride + kk + kb + 16);
  return __builtin_shufflevector(p0[0], p1[0],
                                 0,1,2,3,4,5,6,7,8,9,10,11,12,13,14,15);
}

// ============ Kernel 0a: pack KxN f32 weight into B-fragment f16 order ================
// out[((kt*ntiles + nt)*32 + lane)*16 + j] = W[(kt*32 + 8*(lane>>4) + (j<8?j:j+8))*N + nt*16 + (lane&15)]
__global__ __launch_bounds__(256) void pack_b_kernel(const float* __restrict__ W,
                                                     _Float16* __restrict__ out,
                                                     int K, int N) {
  int tid = blockIdx.x * blockDim.x + threadIdx.x;
  if (tid >= K * N) return;
  int j      = tid & 15;
  int lane   = (tid >> 4) & 31;
  int tile   = tid >> 9;
  int ntiles = N >> 4;
  int kt     = tile / ntiles;
  int nt     = tile - kt * ntiles;
  int Kidx   = kt * 32 + ((lane >> 4) << 3) + (j < 8 ? j : j + 8);
  int Nidx   = nt * 16 + (lane & 15);
  out[tid] = (_Float16)W[(size_t)Kidx * N + Nidx];
}

// ============ Kernel 0b: pack y (8192x512 f32) into A-fragment f16 order ==============
// Layout: [rowtile rt][ktile kt][lane][16]; lane<16: M=lane, K = kb+(j<8?j:j+8), kb=8*(lane>>4)
__global__ __launch_bounds__(256) void pack_y_kernel(const float* __restrict__ y,
                                                     _Float16* __restrict__ yh) {
  int tid = blockIdx.x * blockDim.x + threadIdx.x;   // one packed half per thread
  if (tid >= BB * TT * CC) return;
  int j    = tid & 15;
  int lane = (tid >> 4) & 31;
  int kt   = (tid >> 9) & 15;                        // CC/32 = 16 k-tiles
  int rt   = tid >> 13;                              // row tile (16 rows)
  int m    = lane & 15;
  int kb   = (lane >> 4) * 8;
  int K    = kt * 32 + kb + (j < 8 ? j : j + 8);
  int row  = rt * 16 + m;
  yh[tid] = (_Float16)y[(size_t)row * CC + K];
}

// ===================== Kernel 1: bm = y @ W_proj2  (8192x512 @ 512x128) ===============
// A tile (16x512, f16-packed, 16 KB) staged in LDS once; all 8 waves share it.
__global__ __launch_bounds__(256) void bm_gemm_kernel(const _Float16* __restrict__ yh,
                                                      const _Float16* __restrict__ Wp2h,
                                                      float* __restrict__ bm) {
  __shared__ __align__(16) _Float16 s_a[16 * CC];    // 16 KB, packed fragment order
  int tid  = threadIdx.x;
  int lane = tid & 31;
  int wave = tid >> 5;                               // 8 waves -> 8 N-tiles (MB=128)
  int r0   = blockIdx.x * 16;

  // cooperative A-tile load: 8192 halves = 1024 int4
  {
    const int4* src = (const int4*)(yh + (size_t)blockIdx.x * (16 * CC));
    int4* dst = (int4*)s_a;
#pragma unroll
    for (int i = 0; i < 4; ++i) dst[tid + 256 * i] = src[tid + 256 * i];
  }
  __syncthreads();

  v8f acc = {};
#pragma unroll 4
  for (int kt = 0; kt < CC / 32; ++kt) {
    v16h a = ld_frag_pk(s_a, kt, lane);              // ds_load_b128 x2
    v16h b = ld_frag_pk(Wp2h, kt * 8 + wave, lane);  // global_load_b128 x2
    acc = __builtin_amdgcn_wmma_f32_16x16x32_f16(false, a, false, b, (short)0, acc,
                                                 false, false);
  }
  int n  = lane & 15;
  int mh = (lane >> 4) * 8;
#pragma unroll
  for (int r = 0; r < 8; ++r)
    bm[(size_t)(r0 + mh + r) * MBB + wave * 16 + n] = acc[r];
}

// ===================== Kernel 2: yc = y @ W_proj_comp  (8192x512 @ 512x5) =============
__global__ __launch_bounds__(256) void yc_kernel(const float* __restrict__ y,
                                                 const float* __restrict__ Wpc,
                                                 float* __restrict__ yc) {
  int gid = blockIdx.x * blockDim.x + threadIdx.x;
  if (gid >= BB * TT * CSS) return;
  int k   = gid % CSS;
  int row = gid / CSS;
  const float* yr = y + (size_t)row * CC;
  float acc = 0.f;
#pragma unroll 4
  for (int c = 0; c < CC; c += 4) {
    float4 v = *(const float4*)(yr + c);
    acc += v.x * Wpc[(c + 0) * CSS + k] + v.y * Wpc[(c + 1) * CSS + k]
         + v.z * Wpc[(c + 2) * CSS + k] + v.w * Wpc[(c + 3) * CSS + k];
  }
  yc[gid] = acc;
}

// ==== Kernel 3: fused res -> mix (WMMA) -> gather-weighted out -> final (WMMA) ========
__global__ __launch_bounds__(256) void token_kernel(
    const float* __restrict__ yc, const float* __restrict__ bm,
    const float* __restrict__ W_BASE, const float* __restrict__ W_BIAS,
    const _Float16* __restrict__ W1h, const _Float16* __restrict__ WFh,
    const int* __restrict__ patches, const int* __restrict__ patches_proj,
    float* __restrict__ out) {
  __shared__ __align__(16) float    s_wbase[PP * PSS * CSS];   // 20 KB
  __shared__ __align__(16) _Float16 s_res[16 * PP];            // 2 KB
  __shared__ __align__(16) float    s_mix[16 * PP];            // 4 KB
  __shared__ __align__(16) _Float16 s_out[16 * MBB];           // 4 KB
  __shared__ __align__(16) int      s_pp[16 * PP];             // 4 KB

  int tid  = threadIdx.x;
  int lane = tid & 31;
  int wave = tid >> 5;
  int b    = blockIdx.x / (TT / 16);
  int t0   = (blockIdx.x % (TT / 16)) * 16;

  for (int i = tid; i < PP * PSS * CSS; i += 256) s_wbase[i] = W_BASE[i];
  __syncthreads();

  // ---- res[tok][p] = sum_{s,c} yc[b, patches[t,p,s], c] * W_BASE[p,s,c] + bias[p] ----
#pragma unroll
  for (int i = 0; i < 4; ++i) {
    int w   = tid + 256 * i;
    int tok = w >> 6;
    int p   = w & 63;
    const int*   pr = patches + ((size_t)(t0 + tok) * PP + p) * PSS;
    const float* wb = s_wbase + p * (PSS * CSS);
    float acc = W_BIAS[p];
#pragma unroll
    for (int s = 0; s < PSS; ++s) {
      const float* ycr = yc + ((size_t)b * TT + pr[s]) * CSS;
#pragma unroll
      for (int c = 0; c < CSS; ++c) acc += ycr[c] * wb[s * CSS + c];
    }
    s_res[tok * PP + p] = (_Float16)acc;
  }
  __syncthreads();

  // ---- waves 0..3: mix = res @ W_presoftmax1 (WMMA); waves 4..7: stage gather indices
  if (wave < 4) {
    int q0 = wave * 16;
    v8f acc = {};
#pragma unroll
    for (int kk = 0; kk < PP; kk += 32) {
      v16h a  = ldA_lds(s_res, PP, kk, lane);
      v16h bb = ld_frag_pk(W1h, (kk >> 5) * 4 + wave, lane);
      acc = __builtin_amdgcn_wmma_f32_16x16x32_f16(false, a, false, bb, (short)0, acc,
                                                   false, false);
    }
    int n  = lane & 15;
    int mh = (lane >> 4) * 8;
#pragma unroll
    for (int r = 0; r < 8; ++r)
      s_mix[(mh + r) * PP + q0 + n] = acc[r];
  } else {
    int t = tid - 128;
    const int4* src = (const int4*)(patches_proj + (size_t)t0 * PP);
    int4* dst = (int4*)s_pp;
#pragma unroll
    for (int i = 0; i < 2; ++i) dst[t + 128 * i] = src[t + 128 * i];
  }
  __syncthreads();

  // ---- out[tok][m] = sum_p mix[tok][p] * bm[b, patches_proj[t,p], m]  (L2 gather) ----
  {
    int m0   = (tid & 31) * 4;
    int tokb = tid >> 5;
    const float* bmb = bm + (size_t)b * TT * MBB + m0;
#pragma unroll
    for (int h = 0; h < 2; ++h) {
      int tok = tokb + 8 * h;
      const int4*  pp4 = (const int4*)(s_pp + tok * PP);
      const float* mx  = s_mix + tok * PP;
      float a0 = 0.f, a1 = 0.f, a2 = 0.f, a3 = 0.f;
#pragma unroll 2
      for (int p4 = 0; p4 < PP / 4; ++p4) {
        if (p4 + 2 < PP / 4) {                    // keep ~12 lines in flight
          int4 qf = pp4[p4 + 2];
          __builtin_prefetch(bmb + (size_t)qf.x * MBB, 0, 1);
          __builtin_prefetch(bmb + (size_t)qf.y * MBB, 0, 1);
          __builtin_prefetch(bmb + (size_t)qf.z * MBB, 0, 1);
          __builtin_prefetch(bmb + (size_t)qf.w * MBB, 0, 1);
        }
        int4 q = pp4[p4];
        const float4 v0 = *(const float4*)(bmb + (size_t)q.x * MBB);
        const float4 v1 = *(const float4*)(bmb + (size_t)q.y * MBB);
        const float4 v2 = *(const float4*)(bmb + (size_t)q.z * MBB);
        const float4 v3 = *(const float4*)(bmb + (size_t)q.w * MBB);
        float w0 = mx[4 * p4 + 0], w1 = mx[4 * p4 + 1];
        float w2 = mx[4 * p4 + 2], w3 = mx[4 * p4 + 3];
        a0 += w0 * v0.x + w1 * v1.x + w2 * v2.x + w3 * v3.x;
        a1 += w0 * v0.y + w1 * v1.y + w2 * v2.y + w3 * v3.y;
        a2 += w0 * v0.z + w1 * v1.z + w2 * v2.z + w3 * v3.z;
        a3 += w0 * v0.w + w1 * v1.w + w2 * v2.w + w3 * v3.w;
      }
      s_out[tok * MBB + m0 + 0] = (_Float16)a0;
      s_out[tok * MBB + m0 + 1] = (_Float16)a1;
      s_out[tok * MBB + m0 + 2] = (_Float16)a2;
      s_out[tok * MBB + m0 + 3] = (_Float16)a3;
    }
  }
  __syncthreads();

  // ---- final = out @ W_FINAL  (16x128 @ 128x128), all 8 waves ----
  {
    v8f acc = {};
#pragma unroll
    for (int kk = 0; kk < MBB; kk += 32) {
      v16h a  = ldA_lds(s_out, MBB, kk, lane);
      v16h bb = ld_frag_pk(WFh, (kk >> 5) * 8 + wave, lane);
      acc = __builtin_amdgcn_wmma_f32_16x16x32_f16(false, a, false, bb, (short)0, acc,
                                                   false, false);
    }
    int n  = lane & 15;
    int mh = (lane >> 4) * 8;
#pragma unroll
    for (int r = 0; r < 8; ++r)
      out[((size_t)b * TT + t0 + mh + r) * FDD + wave * 16 + n] = acc[r];
  }
}

// ===================================== launch =========================================
extern "C" void kernel_launch(void* const* d_in, const int* in_sizes, int n_in,
                              void* d_out, int out_size, void* d_ws, size_t ws_size,
                              hipStream_t stream) {
  const float* y       = (const float*)d_in[0];
  const float* W_BASE  = (const float*)d_in[1];
  const float* W_BIAS  = (const float*)d_in[2];
  const float* W_proj2 = (const float*)d_in[3];
  const float* W_pc    = (const float*)d_in[4];
  const float* W1      = (const float*)d_in[5];
  const float* WF      = (const float*)d_in[6];
  const int*   patches      = (const int*)d_in[7];
  const int*   patches_proj = (const int*)d_in[8];
  float* out = (float*)d_out;

  // Workspace layout (16-byte aligned sections)
  float*     bm   = (float*)d_ws;                            // 1,048,576 f32
  float*     yc   = bm + (size_t)BB * TT * MBB;              //    40,960 f32
  _Float16*  Wp2h = (_Float16*)(yc + (size_t)BB * TT * CSS); //    65,536 f16
  _Float16*  W1h  = Wp2h + (size_t)CC * MBB;                 //     4,096 f16
  _Float16*  WFh  = W1h + (size_t)PP * PP;                   //    16,384 f16
  _Float16*  yh   = WFh + (size_t)MBB * FDD;                 // 4,194,304 f16 (A-packed)

  pack_b_kernel<<<(CC * MBB + 255) / 256, 256, 0, stream>>>(W_proj2, Wp2h, CC, MBB);
  pack_b_kernel<<<(PP * PP + 255) / 256, 256, 0, stream>>>(W1, W1h, PP, PP);
  pack_b_kernel<<<(MBB * FDD + 255) / 256, 256, 0, stream>>>(WF, WFh, MBB, FDD);
  pack_y_kernel<<<(BB * TT * CC) / 256, 256, 0, stream>>>(y, yh);

  bm_gemm_kernel<<<(BB * TT) / 16, 256, 0, stream>>>(yh, Wp2h, bm);
  yc_kernel<<<(BB * TT * CSS + 255) / 256, 256, 0, stream>>>(y, W_pc, yc);
  token_kernel<<<BB * (TT / 16), 256, 0, stream>>>(yc, bm, W_BASE, W_BIAS, W1h, WFh,
                                                   patches, patches_proj, out);
}
